// MHAttentionMap_32830730011547
// MI455X (gfx1250) — matrix operand order repaired and checked by reading, not compiled, not run on hardware
//
#include <hip/hip_runtime.h>
#include <hip/hip_bf16.h>

// MHAttentionMap fused kernel set for MI455X (gfx1250, wave32, WMMA + TDM).
//
// Stage 1 (kcast):  k[b][c][hw] f32  ->  khf[b][hw][c] f16   (33.5 MB, L2-resident)
// Stage 2 (aq):     Aq[b,q,n,c] = sum_d (q@Wq^T + bq)[n*32+d] * Wk[n*32+d, c]  (f16, 6.5 MB)
//                   bias[b,q,n] = sum_d qp[n*32+d] * bk[n*32+d]
// Stage 3 (attn):   per (b, q-pair): TDM loads the contiguous 8 KB A-tile into LDS,
//                   S[16 x 4096] = A[16x256] x K_b[256x4096] via v_wmma_f32_16x16x32_f16
//                   (A frags from LDS ds_load_b128, B frags 2x global_load_b128/lane),
//                   + bias, *1/sqrt(32), mask -> -inf, joint softmax over (heads, HW),
//                   single non-temporal store of the 210 MB output.

#define BQ_TOTAL   1600    // B*Q
#define BATCH      16
#define QLEN       100
#define NHEADS     8
#define HDIM       32
#define CDIM       256
#define HWN        4096
#define NORM_FACT  0.17677669529663687f   // 1/sqrt(32)

typedef __attribute__((ext_vector_type(16))) _Float16     v16h;
typedef __attribute__((ext_vector_type(8)))  float        v8f;
typedef __attribute__((ext_vector_type(4)))  unsigned int u32x4;
typedef __attribute__((ext_vector_type(8)))  int          i32x8;
typedef __attribute__((ext_vector_type(4)))  int          i32x4;

#if defined(__has_builtin)
#if __has_builtin(__builtin_amdgcn_tensor_load_to_lds)
#define HAVE_TDM 1
#endif
#endif

struct Frag32B { uint4 lo; uint4 hi; };

__device__ __forceinline__ v16h load_frag16(const void* p0, const void* p1) {
  Frag32B f;
  f.lo = *(const uint4*)p0;
  f.hi = *(const uint4*)p1;
  return __builtin_bit_cast(v16h, f);
}

// ---------------------------------------------------------------------------
// Stage 1: cast+transpose k (f32 [b][256][4096]) -> khf (f16 [b][4096][256]).
// 32x32 tiles through LDS; NT loads since k is read exactly once from HBM.
// ---------------------------------------------------------------------------
__global__ __launch_bounds__(256) void kcast_kernel(const float* __restrict__ k,
                                                    _Float16* __restrict__ khf) {
  __shared__ float tile[32][33];
  const int tx = threadIdx.x, ty = threadIdx.y;       // block (32, 8)
  const int b  = blockIdx.z;
  const int c0 = blockIdx.y * 32;
  const int p0 = blockIdx.x * 32;

  const float* src = k + ((size_t)b * CDIM + c0) * HWN + p0 + tx;
#pragma unroll
  for (int i = 0; i < 4; ++i)
    tile[ty + 8 * i][tx] = __builtin_nontemporal_load(src + (size_t)(ty + 8 * i) * HWN);
  __syncthreads();

  _Float16* dst = khf + ((size_t)b * HWN + p0) * CDIM + c0 + tx;
#pragma unroll
  for (int i = 0; i < 4; ++i)
    dst[(size_t)(ty + 8 * i) * CDIM] = (_Float16)tile[tx][ty + 8 * i];
}

// ---------------------------------------------------------------------------
// Stage 2: Aq (the q-projection folded through Wk) + per-head bias from bk.
// One block per (b,q) row; tiny vs. stage 3 (0.4 GFLOP total).
// ---------------------------------------------------------------------------
__global__ __launch_bounds__(256) void aq_kernel(const float* __restrict__ q,
                                                 const float* __restrict__ Wq,
                                                 const float* __restrict__ bq,
                                                 const float* __restrict__ Wk,
                                                 const float* __restrict__ bk,
                                                 _Float16* __restrict__ Aq,
                                                 float* __restrict__ bias) {
  __shared__ float qrow[CDIM];
  __shared__ float qp[CDIM];
  const int bqi = blockIdx.x;          // 0..1599
  const int t   = threadIdx.x;         // 0..255

  qrow[t] = q[(size_t)bqi * CDIM + t];
  __syncthreads();

  // qp[t] = bq[t] + dot(q_row, Wq[t,:])
  float s = bq[t];
  const float4* wr = (const float4*)(Wq + (size_t)t * CDIM);
#pragma unroll 4
  for (int c4 = 0; c4 < CDIM / 4; ++c4) {
    float4 w = wr[c4];
    const float* qc = &qrow[c4 * 4];
    s += w.x * qc[0] + w.y * qc[1] + w.z * qc[2] + w.w * qc[3];
  }
  qp[t] = s;
  __syncthreads();

  if (t < NHEADS) {
    float bsum = 0.f;
#pragma unroll 8
    for (int d = 0; d < HDIM; ++d) bsum += qp[t * HDIM + d] * bk[t * HDIM + d];
    bias[bqi * NHEADS + t] = bsum;
  }

  // Aq[n][c] = sum_d qp[n*32+d] * Wk[n*32+d][c];  thread handles c=t for each head n.
#pragma unroll
  for (int n = 0; n < NHEADS; ++n) {
    float acc = 0.f;
#pragma unroll 8
    for (int d = 0; d < HDIM; ++d)
      acc += qp[n * HDIM + d] * Wk[(size_t)(n * HDIM + d) * CDIM + t];
    Aq[(size_t)bqi * (NHEADS * CDIM) + n * CDIM + t] = (_Float16)acc;
  }
}

// ---------------------------------------------------------------------------
// Stage 3: fused scores + masked joint softmax.
// Grid (50, 16): one workgroup per (batch, query-pair). 512 threads = 16 waves.
// ---------------------------------------------------------------------------
__global__ __launch_bounds__(512) void attn_kernel(const _Float16* __restrict__ Aq,
                                                   const _Float16* __restrict__ khf,
                                                   const float* __restrict__ biasArr,
                                                   const unsigned char* __restrict__ mask,
                                                   float* __restrict__ out) {
  struct ShmT {
    _Float16 A[16 * CDIM];   // 8 KB A-tile (rows m=0..15 contiguous in Aq)
    float smax[16][2];
    float ssum[16][2];
  };
  __shared__ ShmT shm;

  const int b    = blockIdx.y;
  const int q0   = blockIdx.x * 2;
  const int tid  = threadIdx.x;
  const int lane = tid & 31;
  const int w    = tid >> 5;        // wave 0..15
  const int ln   = lane & 15;
  const int half = lane >> 4;       // 0: q0 rows, 1: q1 rows
  const int colbase = w * 256;

  // ---- Bring the 8 KB A-tile into LDS (TDM 1D descriptor copy) --------------
  const _Float16* aTile = Aq + (size_t)(b * QLEN + q0) * (NHEADS * CDIM);
#ifdef HAVE_TDM
  if (w == 0) {
    const unsigned long long ga = (unsigned long long)(uintptr_t)aTile;
    const unsigned int lds_off  = (unsigned int)(uintptr_t)(void*)&shm;  // flat[31:0] = LDS offset
    u32x4 g0;
    g0[0] = 1u;                                          // count=1, user mode, no gather
    g0[1] = lds_off;                                     // lds_addr
    g0[2] = (unsigned int)(ga & 0xFFFFFFFFu);            // global_addr[31:0]
    g0[3] = (unsigned int)((ga >> 32) & 0x01FFFFFFu)     // global_addr[56:32]
            | (2u << 30);                                // type = 2 ("image")
    i32x8 g1;
    g1[0] = (2 << 16);                                   // data_size = 4 bytes
    g1[1] = (int)(2048u << 16);                          // tensor_dim0 = 2048 elems (8 KB)
    g1[2] = (int)(1u << 16);                             // tensor_dim1 = 1
    g1[3] = (int)(2048u << 16);                          // tile_dim0 = 2048
    g1[4] = 1;                                           // tile_dim1 = 1
    g1[5] = 2048;                                        // tensor_dim0_stride = 2048
    g1[6] = 0;
    g1[7] = 0;
    const i32x4 z4 = {0, 0, 0, 0};
#if __clang_major__ >= 23
    const i32x8 z8 = {0, 0, 0, 0, 0, 0, 0, 0};
    __builtin_amdgcn_tensor_load_to_lds(g0, g1, z4, z4, z8, 0);
#else
    __builtin_amdgcn_tensor_load_to_lds(g0, g1, z4, z4, 0);
#endif
    __builtin_amdgcn_s_wait_tensorcnt(0);
  }
#else
  // Fallback: cooperative copy (512 threads x 16 B = 8 KB).
  ((uint4*)shm.A)[tid] = ((const uint4*)aTile)[tid];
#endif
  __syncthreads();

  // A-fragment source (LDS): lane holds row m=ln -> (q = q0 + m/8, head = m%8).
  const _Float16* aRow = shm.A + ln * CDIM;
  // B-fragment source: lane holds column hw = colbase + t*16 + ln.
  const _Float16* bRow = khf + ((size_t)b * HWN + colbase + ln) * CDIM;
  const unsigned char* mrow = mask + (size_t)b * HWN + colbase + ln;
  __builtin_prefetch(mrow, 0, 1);

  const v8f vzero = {0.f, 0.f, 0.f, 0.f, 0.f, 0.f, 0.f, 0.f};
  v8f acc[16];
#pragma unroll
  for (int t = 0; t < 16; ++t) acc[t] = vzero;

  for (int ks = 0; ks < 8; ++ks) {
    // A k order used here (contiguous, same as B): lanes 0-15 hold K k0..k0+15, lanes 16-31 hold k0+16..k0+31.
    const int ca = ks * 32 + half * 16;
    // B k order used here (contiguous, same as A): lanes 0-15 hold K k0..k0+15, lanes 16-31 hold k0+16..k0+31.
    const int cb = ks * 32 + half * 16;
    v16h afrag = load_frag16(aRow + ca, aRow + ca + 8);    // 2x ds_load_b128
    const char* bp = (const char*)(bRow + cb);
#pragma unroll
    for (int t = 0; t < 16; ++t) {
      v16h bfrag = load_frag16(bp + t * 8192, bp + t * 8192 + 16);  // 2x global_load_b128
      acc[t] = __builtin_amdgcn_wmma_f32_16x16x32_f16(
          false, afrag, false, bfrag, (short)0, acc[t], false, false);
    }
  }

  // Bias per head, scale, mask, running max. VGPR j <-> head j for both halves.
  const int qlane = q0 + half;
  float biasv[8];
#pragma unroll
  for (int j = 0; j < NHEADS; ++j)
    biasv[j] = biasArr[(b * QLEN + qlane) * NHEADS + j];

  const float NEG_INF = -__builtin_inff();
  float m = NEG_INF;
#pragma unroll
  for (int t = 0; t < 16; ++t) {
    const bool mk = mrow[t * 16] != 0;
#pragma unroll
    for (int j = 0; j < 8; ++j) {
      float v = (acc[t][j] + biasv[j]) * NORM_FACT;
      v = mk ? NEG_INF : v;
      acc[t][j] = v;
      m = fmaxf(m, v);
    }
  }
  // Butterfly max within each 16-lane half (xor 1,2,4,8 stays inside the half).
#pragma unroll
  for (int d = 1; d < 16; d <<= 1) m = fmaxf(m, __shfl_xor(m, d, 32));
  if (ln == 0) shm.smax[w][half] = m;
  __syncthreads();
  float gm = NEG_INF;
#pragma unroll
  for (int i = 0; i < 16; ++i) gm = fmaxf(gm, shm.smax[i][half]);

  float s = 0.f;
#pragma unroll
  for (int t = 0; t < 16; ++t) {
#pragma unroll
    for (int j = 0; j < 8; ++j) {
      float e = __expf(acc[t][j] - gm);   // exp(-inf) == 0 handles the mask
      acc[t][j] = e;
      s += e;
    }
  }
#pragma unroll
  for (int d = 1; d < 16; d <<= 1) s += __shfl_xor(s, d, 32);
  if (ln == 0) shm.ssum[w][half] = s;
  __syncthreads();
  float tot = 0.f;
#pragma unroll
  for (int i = 0; i < 16; ++i) tot += shm.ssum[i][half];
  const float inv = 1.0f / tot;

  // out[b][q][head j][hw]; NT stores: output is written once, keep L2 for khf.
  float* obase = out + (size_t)(b * QLEN + qlane) * NHEADS * HWN + colbase + ln;
#pragma unroll
  for (int t = 0; t < 16; ++t) {
#pragma unroll
    for (int j = 0; j < 8; ++j)
      __builtin_nontemporal_store(acc[t][j] * inv,
                                  obase + (size_t)j * HWN + t * 16);
  }
}

// ---------------------------------------------------------------------------
// Launch. Inputs (setup_inputs order): q, k, Wq, bq, Wk, bk, mask(bool8).
// Workspace layout: khf 33,554,432 B | Aq 6,553,600 B | bias 51,200 B (~40.2 MB).
// ---------------------------------------------------------------------------
extern "C" void kernel_launch(void* const* d_in, const int* in_sizes, int n_in,
                              void* d_out, int out_size, void* d_ws, size_t ws_size,
                              hipStream_t stream) {
  const float* q  = (const float*)d_in[0];
  const float* k  = (const float*)d_in[1];
  const float* Wq = (const float*)d_in[2];
  const float* bq = (const float*)d_in[3];
  const float* Wk = (const float*)d_in[4];
  const float* bk = (const float*)d_in[5];
  const unsigned char* mask = (const unsigned char*)d_in[6];

  char* ws = (char*)d_ws;
  _Float16* khf  = (_Float16*)ws;                                 // 16*4096*256*2
  _Float16* Aq   = (_Float16*)(ws + 33554432);                    // 1600*2048*2
  float*    bias = (float*)(ws + 33554432 + 6553600);             // 1600*8*4

  kcast_kernel<<<dim3(HWN / 32, CDIM / 32, BATCH), dim3(32, 8), 0, stream>>>(k, khf);
  aq_kernel<<<dim3(BQ_TOTAL), dim3(256), 0, stream>>>(q, Wq, bq, Wk, bk, Aq, bias);
  attn_kernel<<<dim3(QLEN / 2, BATCH), dim3(512), 0, stream>>>(Aq, khf, bias, mask,
                                                               (float*)d_out);
}
